// Retriever_29042568856164
// MI455X (gfx1250) — matrix operand order repaired and checked by reading, not compiled, not run on hardware
//
#include <hip/hip_runtime.h>
#include <math.h>

// ---------------------------------------------------------------------------
// CDNA5 (gfx1250) fused GNN-retriever forward.
// Heavy path: per-edge bf16 WMMA GEMMs (v_wmma_f32_16x16x32_bf16), wave32,
// 1 wave = one 16-edge M-tile, C-resident register blocking (16 N-tiles of
// f32 accumulators live in VGPRs), weights pre-swizzled into fragment order.
// ---------------------------------------------------------------------------

typedef __attribute__((ext_vector_type(16))) __bf16        v16bf;
typedef __attribute__((ext_vector_type(8)))  float         v8f;
typedef __attribute__((ext_vector_type(4)))  unsigned int  v4u;

union FragBF { v4u q[2]; v16bf v; };

__device__ __forceinline__ unsigned short f2bf(float x) {
    unsigned int u = __float_as_uint(x);
    unsigned int r = u + 0x7FFFu + ((u >> 16) & 1u);   // round-to-nearest-even
    return (unsigned short)(r >> 16);
}

// A fragment 16x32 bf16 (ISA 7.12.2): lane<16 -> M=lane, K = kbase+{0..7,16..23};
// lane>=16 -> M=lane-16, K = kbase+{8..15,24..31}. Two 16B loads per lane.
__device__ __forceinline__ v16bf lda(const unsigned short* p, int kbase, int koff) {
    FragBF f;
    f.q[0] = *(const v4u*)(p + kbase + koff);
    f.q[1] = *(const v4u*)(p + kbase + koff + 16);
    return f.v;
}

// B fragment 32x16 bf16, pre-swizzled: frag (kk,nt) stored as 32 lanes x 16 halves.
__device__ __forceinline__ v16bf ldb(const unsigned short* w, int kk, int nt, int lane) {
    const v4u* q = (const v4u*)(w + ((((size_t)kk * 16 + nt) * 32 + lane) << 4));
    FragBF f; f.q[0] = q[0]; f.q[1] = q[1];
    return f.v;
}

#define WMMA_BF16(a, b, c) \
    __builtin_amdgcn_wmma_f32_16x16x32_bf16(false, (a), false, (b), (short)0, (c), false, false)

// ------------------------------ prep kernels -------------------------------

__global__ void fill_zero_kernel(float* p, long n) {
    long i = (long)blockIdx.x * blockDim.x + threadIdx.x;
    if (i < n) p[i] = 0.f;
}

__global__ void degree_kernel(const int* h, const int* t, float* degf, float* degr, int E) {
    int e = blockIdx.x * blockDim.x + threadIdx.x;
    if (e < E) {
        atomicAdd(&degf[t[e]], 1.f);
        atomicAdd(&degr[h[e]], 1.f);
    }
}

__global__ void finish_deg_kernel(float* degf, float* degr, const float* topic, float* pf, int N) {
    int i = blockIdx.x * blockDim.x + threadIdx.x;
    if (i < N) {
        degf[i] = 1.f / fmaxf(degf[i], 1.f);
        degr[i] = 1.f / fmaxf(degr[i], 1.f);
        pf[(size_t)i * 10 + 0] = topic[(size_t)i * 2 + 0];
        pf[(size_t)i * 10 + 1] = topic[(size_t)i * 2 + 1];
    }
}

__global__ void dde_scatter_kernel(const int* src, const int* dst, const float* pf,
                                   int scol, float* tmp, int E) {
    int e = blockIdx.x * blockDim.x + threadIdx.x;
    if (e < E) {
        int s = src[e], d = dst[e];
        atomicAdd(&tmp[(size_t)d * 2 + 0], pf[(size_t)s * 10 + scol + 0]);
        atomicAdd(&tmp[(size_t)d * 2 + 1], pf[(size_t)s * 10 + scol + 1]);
    }
}

__global__ void dde_scale_kernel(const float* tmp, const float* inv, float* pf, int dcol, int N) {
    int i = blockIdx.x * blockDim.x + threadIdx.x;
    if (i < N) {
        float iv = inv[i];
        pf[(size_t)i * 10 + dcol + 0] = tmp[(size_t)i * 2 + 0] * iv;
        pf[(size_t)i * 10 + dcol + 1] = tmp[(size_t)i * 2 + 1] * iv;
    }
}

__global__ void motif_adj_kernel(const int* ids, const float* wts, float* adj, int E) {
    int e = blockIdx.x * blockDim.x + threadIdx.x;
    if (e < E) {
        int id[4]; float w[4];
        #pragma unroll
        for (int k = 0; k < 4; ++k) { id[k] = ids[(size_t)e * 4 + k]; w[k] = wts[(size_t)e * 4 + k]; }
        #pragma unroll
        for (int i = 0; i < 4; ++i)
            #pragma unroll
            for (int j = 0; j < 4; ++j)
                if (id[i] > 0 && id[j] > 0) {
                    float c = w[i] * w[j];
                    if (c > 0.f) atomicAdd(&adj[id[i] * 17 + id[j]], c);
                }
    }
}

__global__ void adj_norm_kernel(const float* adj, float* adjn) {
    __shared__ float sa[289];
    __shared__ float rinv[17];
    int t = threadIdx.x;
    if (t < 289) {
        int i = t / 17, j = t % 17;
        float v = 0.5f * (adj[i * 17 + j] + adj[j * 17 + i]) + ((i == j) ? 1.f : 0.f);
        if (i == 0 || j == 0) v = (i == 0 && j == 0) ? 1.f : 0.f;
        sa[t] = v;
    }
    __syncthreads();
    if (t < 17) {
        float s = 0.f;
        for (int j = 0; j < 17; ++j) s += sa[t * 17 + j];
        rinv[t] = 1.f / fmaxf(s, 1e-8f);
    }
    __syncthreads();
    if (t < 289) adjn[t] = sa[t] * rinv[t / 17];
}

__global__ void sem_msg_kernel(const float* adjn, const float* me, float* msg) {
    int i = blockIdx.x, j = threadIdx.x;
    float s = 0.f;
    for (int k = 0; k < 17; ++k) s += adjn[i * 17 + k] * me[k * 64 + j];
    msg[i * 64 + j] = s;
}

__global__ void sem_upd_kernel(const float* me, const float* msg, const float* gsw,
                               const float* gsb, const float* gmw, const float* gmb, float* ht) {
    int i = blockIdx.x, j = threadIdx.x;
    float s = gsb[j] + gmb[j];
    for (int k = 0; k < 64; ++k)
        s += me[i * 64 + k] * gsw[k * 64 + j] + msg[i * 64 + k] * gmw[k * 64 + j];
    ht[i * 64 + j] = me[i * 64 + j] + fmaxf(s, 0.f);
}

__global__ void motif_gather_kernel(const float* ht, const int* ids, const float* wts,
                                    unsigned short* out, long NM) {
    long idx = (long)blockIdx.x * blockDim.x + threadIdx.x;
    if (idx < NM) {
        long n = idx >> 6; int j = (int)(idx & 63);
        float s = 0.f;
        #pragma unroll
        for (int k = 0; k < 4; ++k)
            s += ht[(size_t)ids[n * 4 + k] * 64 + j] * wts[n * 4 + k];
        out[idx] = f2bf(s);
    }
}

__global__ void cvt_he_kernel(const float* ee, const float* nte, unsigned short* he,
                              int NTEXT, long NE) {
    long idx = (long)blockIdx.x * blockDim.x + threadIdx.x;
    if (idx < NE) {
        long n = idx >> 8; int c = (int)(idx & 255);
        float v = (n < NTEXT) ? ee[idx] : nte[c];
        he[idx] = f2bf(v);
    }
}

__global__ void cvt_bf_kernel(const float* in, unsigned short* out, long n) {
    long idx = (long)blockIdx.x * blockDim.x + threadIdx.x;
    if (idx < n) out[idx] = f2bf(in[idx]);
}

__global__ void pos_pair_kernel(const float* pf, const int* h, const int* t,
                                unsigned short* pp, int E) {
    long idx = (long)blockIdx.x * blockDim.x + threadIdx.x;
    if (idx < (long)E * 32) {
        long e = idx >> 5; int c = (int)(idx & 31);
        float v = 0.f;
        if (c < 10)       v = pf[(size_t)h[e] * 10 + c];
        else if (c < 20)  v = pf[(size_t)t[e] * 10 + (c - 10)];
        pp[idx] = f2bf(v);
    }
}

// Swizzle W [Ktot,256] fp32 (valid rows Kvalid) into bf16 B-fragment order.
__global__ void swizzle_w_kernel(const float* W, unsigned short* out, int ktiles, int Kvalid) {
    long idx = (long)blockIdx.x * blockDim.x + threadIdx.x;
    long total = (long)ktiles * 16 * 512;
    if (idx < total) {
        int frag = (int)(idx >> 9), within = (int)(idx & 511);
        int lane = within >> 4, j = within & 15;
        int kk = frag >> 4, nt = frag & 15;
        int K = kk * 32 + ((lane < 16) ? 0 : 8) + (j & 7) + ((j >> 3) << 4);   // lanes 0-15: K {0..7,16..23}, 16-31: K {8..15,24..31}
        int n = nt * 16 + (lane & 15);
        float v = (K < Kvalid) ? W[(size_t)K * 256 + n] : 0.f;
        out[idx] = f2bf(v);
    }
}

__global__ void gate_kernel(const float* q, const float* gw, const float* gb, float* gate) {
    __shared__ float s[3];
    int t = threadIdx.x;
    if (t < 3) {
        float a = gb[t];
        for (int k = 0; k < 256; ++k) a += q[k] * gw[k * 3 + t];
        s[t] = a;
    }
    __syncthreads();
    if (t == 0) {
        float mx = fmaxf(s[0], fmaxf(s[1], s[2]));
        float e0 = expf(s[0] - mx), e1 = expf(s[1] - mx), e2 = expf(s[2] - mx);
        float d = e0 + e1 + e2;
        gate[0] = e0 / d; gate[1] = e1 / d; gate[2] = e2 / d;
    }
}

// c0[n] = pb1[n] + q_emb @ pred_w1[0:256, n]  (edge-invariant half of pred_in)
__global__ void c0_kernel(const float* q, const float* pw1, const float* pb1, float* c0) {
    int t = threadIdx.x;
    float a = pb1[t];
    for (int k = 0; k < 256; ++k) a += q[k] * pw1[(size_t)k * 256 + t];
    c0[t] = a;
}

// ------------------------------ fused edge kernel --------------------------

struct EdgeParams {
    const int *h_id, *t_id, *r_id;
    const unsigned short *he, *nm, *tm, *pp, *rel;
    const unsigned short *wnh, *wpos, *wst, *wp1;
    const float *nh_b, *pos_b, *st_b, *c0, *gate, *pw2, *pb2;
    float* out;
    int E;
};

#define SACC_STRIDE 260   // f32 row stride: 260 % 64 == 4 -> rows 0..7 vs 8..15 hit disjoint banks
#define SB16_STRIDE 264   // bf16 row stride: 528B per row, 16B aligned

__global__ __launch_bounds__(128) void edge_fused_kernel(EdgeParams P) {
    __shared__ float          sacc[4][16][SACC_STRIDE];   // per-wave fused f32 accum
    __shared__ unsigned short sb16[4][16][SB16_STRIDE];   // per-wave fused bf16 tile
    const int lane = threadIdx.x & 31;
    const int wave = threadIdx.x >> 5;
    const long tile = (long)blockIdx.x * 4 + wave;
    const long tb = tile * 16;
    if (tb >= P.E) return;

    const int m    = lane & 15;
    const int koff = (lane >= 16) ? 8 : 0;
    const int mrow = (lane >= 16) ? 8 : 0;
    long eid = tb + m; if (eid >= P.E) eid = P.E - 1;

    const int h = P.h_id[eid], t = P.t_id[eid], r = P.r_id[eid];
    const unsigned short* phh = P.he  + (size_t)h   * 256;
    const unsigned short* pht = P.he  + (size_t)t   * 256;
    const unsigned short* pnh = P.nm  + (size_t)h   * 64;
    const unsigned short* pnt = P.nm  + (size_t)t   * 64;
    const unsigned short* ppp = P.pp  + (size_t)eid * 32;
    const unsigned short* prl = P.rel + (size_t)r   * 256;
    const unsigned short* ptm = P.tm  + (size_t)eid * 64;
    const float g0 = P.gate[0], g1 = P.gate[1], g2 = P.gate[2];

    v8f acc[16];   // 16 N-tiles of C resident in registers (128 VGPRs)

    // ================= pass 1: h_nbr  (K = 640, 20 k-tiles) =================
    #pragma unroll
    for (int nt = 0; nt < 16; ++nt) {
        const float b = P.nh_b[nt * 16 + m];
        #pragma unroll
        for (int rr = 0; rr < 8; ++rr) acc[nt][rr] = b;
    }
    for (int kk = 0; kk < 8; ++kk) {
        const v16bf A = lda(phh, kk * 32, koff);
        #pragma unroll
        for (int nt = 0; nt < 16; ++nt)
            acc[nt] = WMMA_BF16(A, ldb(P.wnh, kk, nt, lane), acc[nt]);
    }
    for (int kk = 0; kk < 8; ++kk) {
        const v16bf A = lda(pht, kk * 32, koff);
        #pragma unroll
        for (int nt = 0; nt < 16; ++nt)
            acc[nt] = WMMA_BF16(A, ldb(P.wnh, 8 + kk, nt, lane), acc[nt]);
    }
    for (int kk = 0; kk < 2; ++kk) {
        const v16bf A = lda(pnh, kk * 32, koff);
        #pragma unroll
        for (int nt = 0; nt < 16; ++nt)
            acc[nt] = WMMA_BF16(A, ldb(P.wnh, 16 + kk, nt, lane), acc[nt]);
    }
    for (int kk = 0; kk < 2; ++kk) {
        const v16bf A = lda(pnt, kk * 32, koff);
        #pragma unroll
        for (int nt = 0; nt < 16; ++nt)
            acc[nt] = WMMA_BF16(A, ldb(P.wnh, 18 + kk, nt, lane), acc[nt]);
    }
    #pragma unroll
    for (int nt = 0; nt < 16; ++nt)
        #pragma unroll
        for (int rr = 0; rr < 8; ++rr)
            sacc[wave][mrow + rr][nt * 16 + m] = g0 * fmaxf(acc[nt][rr], 0.f);

    // ================= pass 2: h_pos  (K = 32 padded, 1 k-tile) =============
    {
        const v16bf A = lda(ppp, 0, koff);
        #pragma unroll
        for (int nt = 0; nt < 16; ++nt) {
            const float b = P.pos_b[nt * 16 + m];
            v8f c;
            #pragma unroll
            for (int rr = 0; rr < 8; ++rr) c[rr] = b;
            c = WMMA_BF16(A, ldb(P.wpos, 0, nt, lane), c);
            #pragma unroll
            for (int rr = 0; rr < 8; ++rr)
                sacc[wave][mrow + rr][nt * 16 + m] += g1 * fmaxf(c[rr], 0.f);
        }
    }

    // ================= pass 3: h_str  (K = 320, 10 k-tiles) =================
    #pragma unroll
    for (int nt = 0; nt < 16; ++nt) {
        const float b = P.st_b[nt * 16 + m];
        #pragma unroll
        for (int rr = 0; rr < 8; ++rr) acc[nt][rr] = b;
    }
    for (int kk = 0; kk < 8; ++kk) {
        const v16bf A = lda(prl, kk * 32, koff);
        #pragma unroll
        for (int nt = 0; nt < 16; ++nt)
            acc[nt] = WMMA_BF16(A, ldb(P.wst, kk, nt, lane), acc[nt]);
    }
    for (int kk = 0; kk < 2; ++kk) {
        const v16bf A = lda(ptm, kk * 32, koff);
        #pragma unroll
        for (int nt = 0; nt < 16; ++nt)
            acc[nt] = WMMA_BF16(A, ldb(P.wst, 8 + kk, nt, lane), acc[nt]);
    }
    #pragma unroll
    for (int nt = 0; nt < 16; ++nt)
        #pragma unroll
        for (int rr = 0; rr < 8; ++rr) {
            const float f = sacc[wave][mrow + rr][nt * 16 + m] + g2 * fmaxf(acc[nt][rr], 0.f);
            sb16[wave][mrow + rr][nt * 16 + m] = f2bf(f);
        }

    // ================= pass 4: pred layer (K = 256, 8 k-tiles) ==============
    #pragma unroll
    for (int nt = 0; nt < 16; ++nt) {
        const float c0v = P.c0[nt * 16 + m];
        #pragma unroll
        for (int rr = 0; rr < 8; ++rr) acc[nt][rr] = c0v;
    }
    const unsigned short* pfu = &sb16[wave][m][0];
    for (int kk = 0; kk < 8; ++kk) {
        const v16bf A = lda(pfu, kk * 32, koff);
        #pragma unroll
        for (int nt = 0; nt < 16; ++nt)
            acc[nt] = WMMA_BF16(A, ldb(P.wp1, kk, nt, lane), acc[nt]);
    }
    v8f part;
    #pragma unroll
    for (int rr = 0; rr < 8; ++rr) part[rr] = 0.f;
    #pragma unroll
    for (int nt = 0; nt < 16; ++nt) {
        const float w2v = P.pw2[nt * 16 + m];
        #pragma unroll
        for (int rr = 0; rr < 8; ++rr) part[rr] += fmaxf(acc[nt][rr], 0.f) * w2v;
    }

    // reduce over the 16 N-columns held across lanes (xor stays within each half)
    #pragma unroll
    for (int d = 1; d < 16; d <<= 1) {
        #pragma unroll
        for (int rr = 0; rr < 8; ++rr) part[rr] += __shfl_xor(part[rr], d, 32);
    }
    if (m == 0) {
        const float pb2v = P.pb2[0];
        #pragma unroll
        for (int rr = 0; rr < 8; ++rr) {
            long o = tb + mrow + rr;
            if (o < P.E) P.out[o] = part[rr] + pb2v;
        }
    }
}

// ------------------------------ launch -------------------------------------

extern "C" void kernel_launch(void* const* d_in, const int* in_sizes, int n_in,
                              void* d_out, int out_size, void* d_ws, size_t ws_size,
                              hipStream_t stream) {
    const int*   h_id  = (const int*)  d_in[0];
    const int*   r_id  = (const int*)  d_in[1];
    const int*   t_id  = (const int*)  d_in[2];
    const float* q     = (const float*)d_in[3];
    const float* ee    = (const float*)d_in[4];
    const float* re    = (const float*)d_in[6];
    const float* topic = (const float*)d_in[7];
    const int*   nids  = (const int*)  d_in[8];
    const float* nwts  = (const float*)d_in[9];
    const int*   tids  = (const int*)  d_in[10];
    const float* twts  = (const float*)d_in[11];
    const float* nte   = (const float*)d_in[12];
    const float* me    = (const float*)d_in[13];
    const float* gsw   = (const float*)d_in[14];
    const float* gsb   = (const float*)d_in[15];
    const float* gmw   = (const float*)d_in[16];
    const float* gmb   = (const float*)d_in[17];
    const float* nh_w  = (const float*)d_in[18];
    const float* nh_b  = (const float*)d_in[19];
    const float* pos_w = (const float*)d_in[20];
    const float* pos_b = (const float*)d_in[21];
    const float* st_w  = (const float*)d_in[22];
    const float* st_b  = (const float*)d_in[23];
    const float* gw    = (const float*)d_in[24];
    const float* gb    = (const float*)d_in[25];
    const float* pw1   = (const float*)d_in[26];
    const float* pb1   = (const float*)d_in[27];
    const float* pw2   = (const float*)d_in[28];
    const float* pb2   = (const float*)d_in[29];

    const int E     = in_sizes[0];
    const int N     = in_sizes[7] / 2;
    const int NTEXT = in_sizes[4] / 256;
    const int R     = in_sizes[6] / 256;

    // bump allocator over d_ws (512B aligned)
    char* wp = (char*)d_ws;
    auto alloc = [&](size_t bytes) -> void* {
        void* p = (void*)wp;
        wp += (bytes + 511) & ~(size_t)511;
        return p;
    };
    float* degf = (float*)alloc((size_t)N * 4);
    float* degr = (float*)alloc((size_t)N * 4);
    float* pf   = (float*)alloc((size_t)N * 10 * 4);
    float* tmp  = (float*)alloc((size_t)N * 2 * 4);
    float* adj  = (float*)alloc(289 * 4);
    float* adjn = (float*)alloc(289 * 4);
    float* msg  = (float*)alloc(17 * 64 * 4);
    float* ht   = (float*)alloc(17 * 64 * 4);
    float* gate = (float*)alloc(16);
    float* c0   = (float*)alloc(256 * 4);
    unsigned short* he   = (unsigned short*)alloc((size_t)N * 256 * 2);
    unsigned short* relb = (unsigned short*)alloc((size_t)R * 256 * 2);
    unsigned short* nm   = (unsigned short*)alloc((size_t)N * 64 * 2);
    unsigned short* tm   = (unsigned short*)alloc((size_t)E * 64 * 2);
    unsigned short* pp   = (unsigned short*)alloc((size_t)E * 32 * 2);
    unsigned short* wnh  = (unsigned short*)alloc((size_t)20 * 16 * 512 * 2);
    unsigned short* wpos = (unsigned short*)alloc((size_t)1  * 16 * 512 * 2);
    unsigned short* wst  = (unsigned short*)alloc((size_t)10 * 16 * 512 * 2);
    unsigned short* wp1s = (unsigned short*)alloc((size_t)8  * 16 * 512 * 2);

    auto cdiv = [](long a, long b) { return (int)((a + b - 1) / b); };
    const int T = 256;

    // degrees + topic columns of pos_feat
    fill_zero_kernel<<<cdiv((long)2 * N, T), T, 0, stream>>>(degf, (long)2 * N); // degf+degr adjacent
    degree_kernel<<<cdiv(E, T), T, 0, stream>>>(h_id, t_id, degf, degr, E);
    finish_deg_kernel<<<cdiv(N, T), T, 0, stream>>>(degf, degr, topic, pf, N);

    // 2 forward + 2 reverse DDE rounds -> pos_feat cols 2..9
    const int scol[4] = {0, 2, 0, 6};
    const int dcol[4] = {2, 4, 6, 8};
    for (int rd = 0; rd < 4; ++rd) {
        const bool fwd = (rd < 2);
        fill_zero_kernel<<<cdiv((long)2 * N, T), T, 0, stream>>>(tmp, (long)2 * N);
        dde_scatter_kernel<<<cdiv(E, T), T, 0, stream>>>(
            fwd ? h_id : t_id, fwd ? t_id : h_id, pf, scol[rd], tmp, E);
        dde_scale_kernel<<<cdiv(N, T), T, 0, stream>>>(tmp, fwd ? degf : degr, pf, dcol[rd], N);
    }

    // motif adjacency + semantic GNN (tiny)
    fill_zero_kernel<<<2, T, 0, stream>>>(adj, 289);
    motif_adj_kernel<<<cdiv(E, T), T, 0, stream>>>(tids, twts, adj, E);
    adj_norm_kernel<<<1, 512, 0, stream>>>(adj, adjn);
    sem_msg_kernel<<<17, 64, 0, stream>>>(adjn, me, msg);
    sem_upd_kernel<<<17, 64, 0, stream>>>(me, msg, gsw, gsb, gmw, gmb, ht);

    // bf16 feature tables
    motif_gather_kernel<<<cdiv((long)N * 64, T), T, 0, stream>>>(ht, nids, nwts, nm, (long)N * 64);
    motif_gather_kernel<<<cdiv((long)E * 64, T), T, 0, stream>>>(ht, tids, twts, tm, (long)E * 64);
    cvt_he_kernel<<<cdiv((long)N * 256, T), T, 0, stream>>>(ee, nte, he, NTEXT, (long)N * 256);
    cvt_bf_kernel<<<cdiv((long)R * 256, T), T, 0, stream>>>(re, relb, (long)R * 256);
    pos_pair_kernel<<<cdiv((long)E * 32, T), T, 0, stream>>>(pf, h_id, t_id, pp, E);

    // weight swizzles (B-fragment order) + scalars
    swizzle_w_kernel<<<cdiv((long)20 * 16 * 512, T), T, 0, stream>>>(nh_w, wnh, 20, 640);
    swizzle_w_kernel<<<cdiv((long)1  * 16 * 512, T), T, 0, stream>>>(pos_w, wpos, 1, 20);
    swizzle_w_kernel<<<cdiv((long)10 * 16 * 512, T), T, 0, stream>>>(st_w, wst, 10, 320);
    swizzle_w_kernel<<<cdiv((long)8  * 16 * 512, T), T, 0, stream>>>(pw1 + (size_t)256 * 256, wp1s, 8, 256);
    gate_kernel<<<1, 32, 0, stream>>>(q, gw, gb, gate);
    c0_kernel<<<1, 256, 0, stream>>>(q, pw1, pb1, c0);

    // fused WMMA edge kernel: 4 waves/block, 16 edges/wave, C-resident blocking
    EdgeParams P;
    P.h_id = h_id; P.t_id = t_id; P.r_id = r_id;
    P.he = he; P.nm = nm; P.tm = tm; P.pp = pp; P.rel = relb;
    P.wnh = wnh; P.wpos = wpos; P.wst = wst; P.wp1 = wp1s;
    P.nh_b = nh_b; P.pos_b = pos_b; P.st_b = st_b;
    P.c0 = c0; P.gate = gate; P.pw2 = pw2; P.pb2 = pb2;
    P.out = (float*)d_out; P.E = E;
    edge_fused_kernel<<<cdiv((long)E, 64), 128, 0, stream>>>(P);
}
